// RWKV7Block_48103633715813
// MI455X (gfx1250) — matrix operand order repaired and checked by reading, not compiled, not run on hardware
//
#include <hip/hip_runtime.h>
#include <hip/hip_bf16.h>
#include <math.h>

// ---------------------------------------------------------------------------
// RWKV7 block for MI455X (gfx1250, wave32, WMMA).
//   B=4, T=2048, D=1024, NH=16, HD=64
// Compute-bound on 5 GEMMs (~86 GFLOP) -> bf16 WMMA (v_wmma_f32_16x16x32_bf16)
// with f32 accumulation. Scan collapses to a per-(head,v) scalar linear
// recurrence:  s[v] = w[v]*s[v] + v[v]*Ksum,  out = r*s.
//
// GEMM: 128x128 block tile, 8 waves, each wave a 32x64 strip (8 accumulators,
// 8 WMMA per 32-K step against 12 ds_load_b128). LDS is double-buffered and
// filled with GLOBAL_LOAD_ASYNC_TO_LDS_B128 (ASYNCcnt) when the toolchain
// exposes the builtin; otherwise coalesced b128 global->LDS staging +
// global_prefetch_b8.
//
// Workspace (~213 MB): xb 16MB | wb 10MB | proj 128MB | ksum 0.5MB |
//                      y 32MB | yb 16MB
// ---------------------------------------------------------------------------

#define Bdim 4
#define Tdim 2048
#define Ddim 1024
#define NHdim 16
#define HDdim 64
#define BT (Bdim * Tdim)          // 8192
#define BTD ((size_t)BT * Ddim)   // 8388608 elements

typedef __attribute__((ext_vector_type(16))) __bf16 v16bf;
typedef __attribute__((ext_vector_type(8)))  __bf16 v8bf;
typedef __attribute__((ext_vector_type(8)))  float  v8f;
typedef __attribute__((ext_vector_type(4)))  int    v4i;

// AS-qualified b128 pointee types for the async-to-LDS builtin
// (param type per hipcc diagnostic: int __vector_size(16) in AS1 / AS3)
typedef __attribute__((address_space(1))) v4i as1_v4i;
typedef __attribute__((address_space(3))) v4i as3_v4i;

// ---- CDNA5 async global->LDS path (guarded; falls back to sync staging) ----
#if defined(__HIP_DEVICE_COMPILE__) && defined(__gfx1250__)
#if __has_builtin(__builtin_amdgcn_global_load_async_to_lds_b128)
#define HAVE_ASYNC_LDS 1
#endif
#endif

#ifdef HAVE_ASYNC_LDS
#if __has_builtin(__builtin_amdgcn_s_wait_asynccnt)
#define WAIT_ASYNC() __builtin_amdgcn_s_wait_asynccnt(0)
#else
#define WAIT_ASYNC() asm volatile("s_wait_asynccnt 0" ::: "memory")
#endif
#define ASYNC_B128(gp, lp)                                                   \
    __builtin_amdgcn_global_load_async_to_lds_b128(                          \
        (as1_v4i*)(void*)(gp), (as3_v4i*)(void*)(lp), 0, 0)
#else
#define WAIT_ASYNC()
#endif

// ---------------------------------------------------------------------------
// f32 -> bf16 conversion
// ---------------------------------------------------------------------------
__global__ void cvt_f32_bf16(const float* __restrict__ in,
                             __bf16* __restrict__ out, int n) {
    int i = blockIdx.x * blockDim.x + threadIdx.x;
    if (i < n) out[i] = (__bf16)in[i];
}

// ---------------------------------------------------------------------------
// WMMA GEMM:  out[M,N] = X[M,K] @ W[N,K]^T   (F.linear, both row-major)
// mode==1: fused projection epilogue; N==4096 is [Wr;Wk;Wv;Ww] concatenated,
//          quarter p = n>>10 : p==0 -> sigmoid (r), p==3 -> 0.99*sigmoid (w);
//          output written as 4 contiguous [M,1024] planes.
// mode==0: plain f32 store out[m*N+n].
// ---------------------------------------------------------------------------
__global__ __launch_bounds__(256)
void wmma_gemm_bf16(const __bf16* __restrict__ X, const __bf16* __restrict__ W,
                    float* __restrict__ out, int M, int N, int K, int mode) {
    // double-buffered tiles, padded stride 40 bf16 (80B) keeps 16B alignment
    // and spreads LDS banks.  2*2*128*40*2 = 40 KB.
    __shared__ __align__(16) __bf16 Xs[2][128][40];
    __shared__ __align__(16) __bf16 Ws[2][128][40];

    const int m0 = blockIdx.x * 128;
    const int n0 = blockIdx.y * 128;

    const int tid   = threadIdx.x;
    const int lane  = tid & 31;
    const int wave  = tid >> 5;
    const int mwave = wave & 3;   // 32-row band
    const int nwave = wave >> 2;  // 64-col band
    const int lr    = lane & 15;
    const int kh    = (lane < 16) ? 0 : 8;   // A-fragment K pattern
    const int ko    = (lane < 16) ? 0 : 8;   // B-fragment K pattern

    v8f acc[2][4] = {};

    // stage one 128x32 X tile + 128x32 W tile into LDS buffer `buf`
    auto stage = [&](int buf, int kb) {
#pragma unroll
        for (int i = 0; i < 2; ++i) {
            int idx = tid + i * 256;
            int row = idx >> 2, ch = idx & 3;
            const __bf16* gx = &X[(size_t)(m0 + row) * K + kb + ch * 8];
            const __bf16* gw = &W[(size_t)(n0 + row) * K + kb + ch * 8];
            __bf16* lx = &Xs[buf][row][ch * 8];
            __bf16* lw = &Ws[buf][row][ch * 8];
#ifdef HAVE_ASYNC_LDS
            ASYNC_B128(gx, lx);
            ASYNC_B128(gw, lw);
#else
            *(v8bf*)lx = *(const v8bf*)gx;
            *(v8bf*)lw = *(const v8bf*)gw;
            __builtin_prefetch(gx + 32, 0, 1);   // global_prefetch_b8, next K tile
            __builtin_prefetch(gw + 32, 0, 1);
#endif
        }
    };

    stage(0, 0);
    int buf = 0;
    for (int kb = 0; kb < K; kb += 32) {
        WAIT_ASYNC();          // our async copies into `buf` have landed
        __syncthreads();       // ... and everyone else's too
        if (kb + 32 < K) stage(buf ^ 1, kb + 32);  // fill other buffer (async)

        // ---- A fragments: rows mwave*32 + {0,16} + lr ----
        v16bf a[2];
#pragma unroll
        for (int mi = 0; mi < 2; ++mi) {
            int row = mwave * 32 + mi * 16 + lr;
            v8bf alo = *(const v8bf*)&Xs[buf][row][kh];
            v8bf ahi = *(const v8bf*)&Xs[buf][row][16 + kh];
            a[mi] = __builtin_shufflevector(alo, ahi,
                0, 1, 2, 3, 4, 5, 6, 7, 8, 9, 10, 11, 12, 13, 14, 15);
        }
#pragma unroll
        for (int j = 0; j < 4; ++j) {
            // B[k,n] = W[n,k]; lane<16 holds K0..7,16..23 of column lr
            int wr = nwave * 64 + j * 16 + lr;
            v8bf blo = *(const v8bf*)&Ws[buf][wr][ko];
            v8bf bhi = *(const v8bf*)&Ws[buf][wr][16 + ko];
            v16bf b = __builtin_shufflevector(blo, bhi,
                0, 1, 2, 3, 4, 5, 6, 7, 8, 9, 10, 11, 12, 13, 14, 15);
            acc[0][j] = __builtin_amdgcn_wmma_f32_16x16x32_bf16(
                false, a[0], false, b, (short)0, acc[0][j], false, false);
            acc[1][j] = __builtin_amdgcn_wmma_f32_16x16x32_bf16(
                false, a[1], false, b, (short)0, acc[1][j], false, false);
        }
        __syncthreads();       // done reading `buf` before it is refilled
        buf ^= 1;
    }

    // ---- epilogue: C/D layout VGPR i -> M = i + (lane>=16 ? 8 : 0), N = lr --
#pragma unroll
    for (int mi = 0; mi < 2; ++mi) {
        int rbase = m0 + mwave * 32 + mi * 16 + ((lane >= 16) ? 8 : 0);
#pragma unroll
        for (int j = 0; j < 4; ++j) {
            int gn = n0 + nwave * 64 + j * 16 + lr;
#pragma unroll
            for (int i = 0; i < 8; ++i) {
                int gm = rbase + i;
                float val = acc[mi][j][i];
                if (mode) {
                    int p = gn >> 10;      // projection id (uniform per block)
                    int c = gn & 1023;
                    if (p == 0)      val = 1.0f / (1.0f + __expf(-val));   // r
                    else if (p == 3) val = 0.99f / (1.0f + __expf(-val));  // w
                    out[((size_t)p * M + gm) * 1024 + c] = val;
                } else {
                    out[(size_t)gm * N + gn] = val;
                }
            }
        }
    }
}

// ---------------------------------------------------------------------------
// Ksum[b,t,h] = sum over hd of k[b,t,h,:]   (one wave32 per head-token)
// ---------------------------------------------------------------------------
__global__ void ksum_kernel(const float* __restrict__ k, float* __restrict__ ksum) {
    int gwave = (blockIdx.x * blockDim.x + threadIdx.x) >> 5;
    int lane = threadIdx.x & 31;
    if (gwave >= BT * NHdim) return;
    const float* kp = k + (size_t)gwave * HDdim;
    float s = kp[lane] + kp[lane + 32];
#pragma unroll
    for (int off = 16; off; off >>= 1) s += __shfl_xor(s, off, 32);
    if (lane == 0) ksum[gwave] = s;
}

// ---------------------------------------------------------------------------
// Sequential scan, collapsed state: s[v] = w[v]*s[v] + v[v]*Ksum ; y = r*s
// One block per (b,head), 64 threads. Register-pipelined (t+1 loads overlap
// the t-step FMA).
// ---------------------------------------------------------------------------
__global__ __launch_bounds__(64)
void scan_kernel(const float* __restrict__ r, const float* __restrict__ w,
                 const float* __restrict__ v, const float* __restrict__ ksum,
                 float* __restrict__ y) {
    int bh = blockIdx.x;           // 0 .. B*NH-1
    int b = bh / NHdim, h = bh % NHdim;
    int tid = threadIdx.x;         // v index 0..63

    size_t base    = (((size_t)b * Tdim) * NHdim + h) * HDdim + tid;
    size_t strideT = (size_t)NHdim * HDdim;   // 1024
    size_t kbase   = (size_t)b * Tdim * NHdim + h;

    float s = 0.0f;
    float wv = w[base], vv = v[base], rv = r[base], ks = ksum[kbase];

    for (int t = 0; t < Tdim; ++t) {
        float wn = 0.f, vn = 0.f, rn = 0.f, kn = 0.f;
        if (t + 1 < Tdim) {
            size_t idx = base + (size_t)(t + 1) * strideT;
            wn = w[idx]; vn = v[idx]; rn = r[idx];
            kn = ksum[kbase + (size_t)(t + 1) * NHdim];
        }
        s = fmaf(vv, ks, wv * s);
        y[base + (size_t)t * strideT] = rv * s;
        wv = wn; vv = vn; rv = rn; ks = kn;
    }
}

// ---------------------------------------------------------------------------
// RMSNorm over D=1024, eps = FLT_EPSILON (matches jnp.finfo(f32).eps), output
// bf16 ready for the Wo WMMA GEMM. One 256-thread block per token.
// ---------------------------------------------------------------------------
__global__ __launch_bounds__(256)
void rmsnorm_kernel(const float* __restrict__ y, __bf16* __restrict__ yb) {
    __shared__ float red[8];
    int tok = blockIdx.x;
    int tid = threadIdx.x;
    const float* row = y + (size_t)tok * Ddim;

    float vals[4];
    float ss = 0.f;
#pragma unroll
    for (int i = 0; i < 4; ++i) {
        float x = row[tid + i * 256];
        vals[i] = x;
        ss += x * x;
    }
#pragma unroll
    for (int off = 16; off; off >>= 1) ss += __shfl_xor(ss, off, 32);
    if ((tid & 31) == 0) red[tid >> 5] = ss;
    __syncthreads();
    float tot = 0.f;
#pragma unroll
    for (int i = 0; i < 8; ++i) tot += red[i];
    float scale = rsqrtf(tot * (1.0f / Ddim) + 1.1920928955078125e-7f);
#pragma unroll
    for (int i = 0; i < 4; ++i)
        yb[(size_t)tok * Ddim + tid + i * 256] = (__bf16)(vals[i] * scale);
}

// ---------------------------------------------------------------------------
// Host launcher
// ---------------------------------------------------------------------------
extern "C" void kernel_launch(void* const* d_in, const int* in_sizes, int n_in,
                              void* d_out, int out_size, void* d_ws, size_t ws_size,
                              hipStream_t stream) {
    const float* x  = (const float*)d_in[0];
    const float* Wp[5] = { (const float*)d_in[1], (const float*)d_in[2],
                           (const float*)d_in[3], (const float*)d_in[4],
                           (const float*)d_in[5] };   // Wr,Wk,Wv,Ww,Wo
    float* outp = (float*)d_out;

    char* ws = (char*)d_ws;
    size_t off = 0;
    __bf16* xb   = (__bf16*)(ws + off); off += BTD * 2;                 // 16 MB
    __bf16* wb   = (__bf16*)(ws + off); off += 5ull * Ddim * Ddim * 2;  // 10 MB
    float*  proj = (float*)(ws + off);  off += 4ull * BTD * 4;          // 128 MB
    float*  ksum = (float*)(ws + off);  off += (size_t)BT * NHdim * 4;  // 0.5 MB
    float*  yf   = (float*)(ws + off);  off += BTD * 4;                 // 32 MB
    __bf16* yb   = (__bf16*)(ws + off); off += BTD * 2;                 // 16 MB

    // 1) convert activations + weights to bf16 (weights packed contiguously)
    cvt_f32_bf16<<<(int)(BTD / 256), 256, 0, stream>>>(x, xb, (int)BTD);
    for (int i = 0; i < 5; ++i)
        cvt_f32_bf16<<<(Ddim * Ddim) / 256, 256, 0, stream>>>(
            Wp[i], wb + (size_t)i * Ddim * Ddim, Ddim * Ddim);

    // 2) fused r/k/v/w projection: one 8192 x 4096 x 1024 WMMA GEMM
    wmma_gemm_bf16<<<dim3(BT / 128, 4096 / 128), 256, 0, stream>>>(
        xb, wb, proj, BT, 4096, Ddim, /*mode=*/1);

    // 3) Ksum reduction (k plane is proj quarter 1)
    {
        int waves = BT * NHdim;                      // 131072
        ksum_kernel<<<(waves * 32) / 256, 256, 0, stream>>>(proj + 1 * BTD, ksum);
    }

    // 4) collapsed time recurrence (r=q0, v=q2, w=q3)
    scan_kernel<<<Bdim * NHdim, 64, 0, stream>>>(
        proj + 0 * BTD, proj + 3 * BTD, proj + 2 * BTD, ksum, yf);

    // 5) RMSNorm -> bf16
    rmsnorm_kernel<<<BT, 256, 0, stream>>>(yf, yb);

    // 6) output projection with Wo -> f32 d_out
    wmma_gemm_bf16<<<dim3(BT / 128, Ddim / 128), 256, 0, stream>>>(
        yb, wb + 4ull * Ddim * Ddim, outp, BT, Ddim, Ddim, /*mode=*/0);
}
